// Attention_8924942041930
// MI455X (gfx1250) — compile-verified, matrix operand order checked by reading, not run on hardware
//
#include <hip/hip_runtime.h>

#define DIM      768
#define N3       2304
#define NSEQ     384
#define BATCH    64
#define NHEADS   12
#define HD       64
#define MROWS    (BATCH*NSEQ)   /* 24576 */
#define QK_SCALE 0.125f         /* 64^-0.5 */

typedef __attribute__((ext_vector_type(16))) _Float16 v16h;
typedef __attribute__((ext_vector_type(8)))  _Float16 v8h;
typedef __attribute__((ext_vector_type(4)))  _Float16 v4h;
typedef __attribute__((ext_vector_type(8)))  float    v8f;
typedef __attribute__((ext_vector_type(4)))  float    v4f;

__device__ __forceinline__ v16h concat8(v8h lo, v8h hi) {
  return __builtin_shufflevector(lo, hi, 0,1,2,3,4,5,6,7,8,9,10,11,12,13,14,15);
}

__device__ __forceinline__ v8f wmma_f16(v16h a, v16h b, v8f c) {
  // v_wmma_f32_16x16x32_f16: D = A(16x32 f16) * B(32x16 f16) + C(16x16 f32)
  return __builtin_amdgcn_wmma_f32_16x16x32_f16(false, a, false, b, (short)0, c,
                                                false, false);
}

// 16x32 f16 WMMA A-fragment (B-fragment when source is column-major) from
// row-major memory, leading dimension `ld` halfs.  ISA layout: lanes 0-15
// row r hold K {0..7},{16..23}; lanes 16-31 hold K {8..15},{24..31}.
__device__ __forceinline__ v16h load_frag_f16(const _Float16* base, int ld, int lane) {
  int r  = lane & 15;
  int ko = (lane < 16) ? 0 : 8;
  const _Float16* p = base + (size_t)r * ld + ko;
  v8h lo = *(const v8h*)(p);
  v8h hi = *(const v8h*)(p + 16);
  return concat8(lo, hi);
}

// ---------------------------------------------------------------------------
// Kernel 0a: fp32 -> f16 bulk convert (x).  b128 in, b64 out, coalesced.
// ---------------------------------------------------------------------------
__global__ __launch_bounds__(256)
void cvt_f16_kernel(const float* __restrict__ src, _Float16* __restrict__ dst) {
  size_t i = ((size_t)blockIdx.x * 256 + threadIdx.x) * 4;
  v4f v = *(const v4f*)(src + i);
  v4h h;
  h[0] = (_Float16)v[0]; h[1] = (_Float16)v[1];
  h[2] = (_Float16)v[2]; h[3] = (_Float16)v[3];
  *(v4h*)(dst + i) = h;
}

// ---------------------------------------------------------------------------
// Kernel 0b: one-shot transpose+convert: src fp32 [R][C] -> dst f16 [C][R].
//   32x32 LDS tile, both global accesses coalesced.  grid (C/32, R/32), 256 thr.
// ---------------------------------------------------------------------------
__global__ __launch_bounds__(256)
void transpose_cvt_kernel(const float* __restrict__ src, _Float16* __restrict__ dst,
                          int R, int C) {
  __shared__ float tile[32][33];
  const int bx = blockIdx.x * 32;          // column offset in src
  const int by = blockIdx.y * 32;          // row offset in src
  const int tx = threadIdx.x & 31;
  const int ty = threadIdx.x >> 5;         // 0..7
  #pragma unroll
  for (int i = 0; i < 32; i += 8)
    tile[ty + i][tx] = src[(size_t)(by + ty + i) * C + bx + tx];
  __syncthreads();
  #pragma unroll
  for (int i = 0; i < 32; i += 8)
    dst[(size_t)(bx + ty + i) * R + by + tx] = (_Float16)tile[tx][ty + i];
}

// ---------------------------------------------------------------------------
// Kernel 1: qkv = xh @ wqkvT^T, scattered into q (pre-scaled), k, v^T (f16).
//   No LDS: A and B fragments load straight from f16 global (L2 resident).
//   8 waves as 4x2; each wave owns a 32x64 tile (8 accumulators).
//   grid (MROWS/128, N3/128) = (192, 18), 256 threads.
// ---------------------------------------------------------------------------
__global__ __launch_bounds__(256)
void qkv_gemm_kernel(const _Float16* __restrict__ xh,
                     const _Float16* __restrict__ wT,   // [N3][DIM] f16
                     _Float16* __restrict__ qh, _Float16* __restrict__ kh,
                     _Float16* __restrict__ vT) {
  const int lane = threadIdx.x & 31;
  const int wave = threadIdx.x >> 5;
  const int wm   = wave >> 1;                        // 0..3
  const int wn   = wave & 1;                         // 0..1
  const int m0   = blockIdx.x * 128 + wm * 32;
  const int n0   = blockIdx.y * 128 + wn * 64;

  const _Float16* ap0 = xh + (size_t)m0 * DIM;
  const _Float16* ap1 = xh + (size_t)(m0 + 16) * DIM;
  const _Float16* bp  = wT + (size_t)n0 * DIM;

  v8f acc[2][4];
  #pragma unroll
  for (int mi = 0; mi < 2; ++mi)
    #pragma unroll
    for (int t = 0; t < 4; ++t) acc[mi][t] = (v8f){};

  for (int k0 = 0; k0 < DIM; k0 += 32) {
    v16h a0 = load_frag_f16(ap0 + k0, DIM, lane);
    v16h a1 = load_frag_f16(ap1 + k0, DIM, lane);
    #pragma unroll
    for (int t = 0; t < 4; ++t) {
      v16h bf = load_frag_f16(bp + (size_t)(t * 16) * DIM + k0, DIM, lane);
      acc[0][t] = wmma_f16(a0, bf, acc[0][t]);
      acc[1][t] = wmma_f16(a1, bf, acc[1][t]);
    }
  }

  // Scatter: D layout — vgpr j: (row j, lanes 0-15) / (row j+8, lanes 16-31)
  #pragma unroll
  for (int mi = 0; mi < 2; ++mi) {
    const int rbase = m0 + mi * 16 + ((lane < 16) ? 0 : 8);
    #pragma unroll
    for (int t = 0; t < 4; ++t) {
      int c     = n0 + t * 16 + (lane & 15);
      int which = c / DIM;                 // uniform across the 16-lane span
      int rem   = c - which * DIM;
      int h     = rem >> 6;
      int d     = rem & 63;
      #pragma unroll
      for (int j = 0; j < 8; ++j) {
        int m = rbase + j;
        int b = m / NSEQ;
        int n = m - b * NSEQ;
        size_t bh = (size_t)b * NHEADS + h;
        float v = acc[mi][t][j];
        if (which == 0) {
          qh[(bh * NSEQ + n) * HD + d] = (_Float16)(v * QK_SCALE);
        } else if (which == 1) {
          kh[(bh * NSEQ + n) * HD + d] = (_Float16)v;
        } else {
          vT[(bh * HD + d) * NSEQ + n] = (_Float16)v;   // transposed for PV
        }
      }
    }
  }
}

// ---------------------------------------------------------------------------
// Kernel 2: flash attention. Output row n = softmax(q_n . K_range) @ V_range
//   rows 64..127 attend keys [64,128); all others attend [0,384).
//   128 threads = 4 waves, one 16-query tile per wave. grid = B*H*6.
// ---------------------------------------------------------------------------
__global__ __launch_bounds__(128)
void attention_kernel(const _Float16* __restrict__ qh,
                      const _Float16* __restrict__ kh,
                      const _Float16* __restrict__ vT,
                      _Float16* __restrict__ attn_out) {
  __shared__ __align__(16) _Float16 P[4][16 * 32];  // per-wave prob tile
  const int tid  = threadIdx.x;
  const int lane = tid & 31;
  const int wave = tid >> 5;

  const int bh = blockIdx.x / 6;
  const int qt = (blockIdx.x % 6) * 4 + wave;       // 0..23
  const int b  = bh / NHEADS;
  const int h  = bh - b * NHEADS;

  const _Float16* qbase = qh + (size_t)bh * NSEQ * HD;
  const _Float16* kbase = kh + (size_t)bh * NSEQ * HD;
  const _Float16* vbase = vT + (size_t)bh * HD * NSEQ;

  const int q0 = qt * 16;
  int kbeg, kend;
  if (qt >= 4 && qt < 8) { kbeg = 64; kend = 128; }   // inhibition block
  else                   { kbeg = 0;  kend = NSEQ; }  // mixed block

  v16h qf0 = load_frag_f16(qbase + (size_t)q0 * HD +  0, HD, lane);
  v16h qf1 = load_frag_f16(qbase + (size_t)q0 * HD + 32, HD, lane);

  float mrow[8], lrow[8];
  #pragma unroll
  for (int j = 0; j < 8; ++j) { mrow[j] = -1e30f; lrow[j] = 0.0f; }
  v8f o0 = {}, o1 = {}, o2 = {}, o3 = {};

  _Float16* pl = &P[wave][0];

  for (int kk = kbeg; kk < kend; kk += 32) {
    // S = Q K^T for 32 keys (two 16-key column tiles)
    v16h ka0 = load_frag_f16(kbase + (size_t)kk        * HD,      HD, lane);
    v16h ka1 = load_frag_f16(kbase + (size_t)kk        * HD + 32, HD, lane);
    v16h kb0 = load_frag_f16(kbase + (size_t)(kk + 16) * HD,      HD, lane);
    v16h kb1 = load_frag_f16(kbase + (size_t)(kk + 16) * HD + 32, HD, lane);
    v8f s0 = {}; s0 = wmma_f16(qf0, ka0, s0); s0 = wmma_f16(qf1, ka1, s0);
    v8f s1 = {}; s1 = wmma_f16(qf0, kb0, s1); s1 = wmma_f16(qf1, kb1, s1);

    // Online softmax. Row j lives in lanes 0-15 (element j) / lanes 16-31
    // (element j -> row j+8); xor-shuffles over masks 1..8 reduce each half.
    #pragma unroll
    for (int j = 0; j < 8; ++j) {
      float v = fmaxf(s0[j], s1[j]);
      #pragma unroll
      for (int msk = 1; msk < 16; msk <<= 1) v = fmaxf(v, __shfl_xor(v, msk, 32));
      float nm = fmaxf(mrow[j], v);
      float sc = __expf(mrow[j] - nm);
      mrow[j] = nm;
      float p0 = __expf(s0[j] - nm);
      float p1 = __expf(s1[j] - nm);
      float rs = p0 + p1;
      #pragma unroll
      for (int msk = 1; msk < 16; msk <<= 1) rs += __shfl_xor(rs, msk, 32);
      lrow[j] = lrow[j] * sc + rs;
      o0[j] = o0[j] * sc; o1[j] = o1[j] * sc;
      o2[j] = o2[j] * sc; o3[j] = o3[j] * sc;
      int row = (lane < 16) ? j : (j + 8);
      pl[row * 32 +      (lane & 15)] = (_Float16)p0;
      pl[row * 32 + 16 + (lane & 15)] = (_Float16)p1;
    }
    asm volatile("s_wait_dscnt 0" ::: "memory");  // P tile visible (same wave)

    // O += P (16x32) @ V (32x64): B-fragments from vT rows (contiguous keys)
    v16h pf  = load_frag_f16(pl, 32, lane);
    v16h bv0 = load_frag_f16(vbase + (size_t) 0 * NSEQ + kk, NSEQ, lane);
    v16h bv1 = load_frag_f16(vbase + (size_t)16 * NSEQ + kk, NSEQ, lane);
    v16h bv2 = load_frag_f16(vbase + (size_t)32 * NSEQ + kk, NSEQ, lane);
    v16h bv3 = load_frag_f16(vbase + (size_t)48 * NSEQ + kk, NSEQ, lane);
    o0 = wmma_f16(pf, bv0, o0);
    o1 = wmma_f16(pf, bv1, o1);
    o2 = wmma_f16(pf, bv2, o2);
    o3 = wmma_f16(pf, bv3, o3);
  }

  // Finalize and store f16 into [B, N, DIM] (= [b, n, h*64 + d])
  const int rbase = q0 + ((lane < 16) ? 0 : 8);
  const int dbase = lane & 15;
  #pragma unroll
  for (int j = 0; j < 8; ++j) {
    float inv = 1.0f / lrow[j];
    size_t o = ((size_t)b * NSEQ + (rbase + j)) * DIM + h * HD;
    attn_out[o +  0 + dbase] = (_Float16)(o0[j] * inv);
    attn_out[o + 16 + dbase] = (_Float16)(o1[j] * inv);
    attn_out[o + 32 + dbase] = (_Float16)(o2[j] * inv);
    attn_out[o + 48 + dbase] = (_Float16)(o3[j] * inv);
  }
}

// ---------------------------------------------------------------------------
// Kernel 3: out = attn @ w_proj + b_proj (fp32 out).  Same no-LDS tiling.
//   grid (MROWS/128, DIM/128) = (192, 6), 256 threads.
// ---------------------------------------------------------------------------
__global__ __launch_bounds__(256)
void proj_gemm_kernel(const _Float16* __restrict__ A,
                      const _Float16* __restrict__ wT,   // [DIM][DIM] f16
                      const float* __restrict__ bias, float* __restrict__ out) {
  const int lane = threadIdx.x & 31;
  const int wave = threadIdx.x >> 5;
  const int wm   = wave >> 1;
  const int wn   = wave & 1;
  const int m0   = blockIdx.x * 128 + wm * 32;
  const int n0   = blockIdx.y * 128 + wn * 64;

  const _Float16* ap0 = A + (size_t)m0 * DIM;
  const _Float16* ap1 = A + (size_t)(m0 + 16) * DIM;
  const _Float16* bp  = wT + (size_t)n0 * DIM;

  v8f acc[2][4];
  #pragma unroll
  for (int mi = 0; mi < 2; ++mi)
    #pragma unroll
    for (int t = 0; t < 4; ++t) acc[mi][t] = (v8f){};

  for (int k0 = 0; k0 < DIM; k0 += 32) {
    v16h a0 = load_frag_f16(ap0 + k0, DIM, lane);
    v16h a1 = load_frag_f16(ap1 + k0, DIM, lane);
    #pragma unroll
    for (int t = 0; t < 4; ++t) {
      v16h bf = load_frag_f16(bp + (size_t)(t * 16) * DIM + k0, DIM, lane);
      acc[0][t] = wmma_f16(a0, bf, acc[0][t]);
      acc[1][t] = wmma_f16(a1, bf, acc[1][t]);
    }
  }

  #pragma unroll
  for (int mi = 0; mi < 2; ++mi) {
    const int rbase = m0 + mi * 16 + ((lane < 16) ? 0 : 8);
    #pragma unroll
    for (int t = 0; t < 4; ++t) {
      int c = n0 + t * 16 + (lane & 15);
      float bb = bias[c];
      #pragma unroll
      for (int j = 0; j < 8; ++j) {
        out[(size_t)(rbase + j) * DIM + c] = acc[mi][t][j] + bb;
      }
    }
  }
}

// ---------------------------------------------------------------------------
extern "C" void kernel_launch(void* const* d_in, const int* in_sizes, int n_in,
                              void* d_out, int out_size, void* d_ws, size_t ws_size,
                              hipStream_t stream) {
  (void)in_sizes; (void)n_in; (void)out_size; (void)ws_size;
  const float* x     = (const float*)d_in[0];
  const float* wqkv  = (const float*)d_in[1];
  const float* wproj = (const float*)d_in[2];
  const float* bproj = (const float*)d_in[3];
  float*       out   = (float*)d_out;

  char* ws = (char*)d_ws;
  const size_t hbuf = (size_t)BATCH * NHEADS * NSEQ * HD * sizeof(_Float16); // 37.75 MB
  size_t off = 0;
  _Float16* xh     = (_Float16*)(ws + off); off += hbuf;   // [MROWS][DIM]
  _Float16* qh     = (_Float16*)(ws + off); off += hbuf;
  _Float16* kh     = (_Float16*)(ws + off); off += hbuf;
  _Float16* vT     = (_Float16*)(ws + off); off += hbuf;
  _Float16* ao     = (_Float16*)(ws + off); off += hbuf;   // [B,N,DIM]
  _Float16* wqkvT  = (_Float16*)(ws + off); off += (size_t)N3 * DIM * sizeof(_Float16);
  _Float16* wprojT = (_Float16*)(ws + off); off += (size_t)DIM * DIM * sizeof(_Float16);

  // one-shot layout prep (all L2-resident afterwards)
  cvt_f16_kernel<<<(MROWS * DIM) / (256 * 4), 256, 0, stream>>>(x, xh);
  { dim3 g(N3 / 32, DIM / 32);
    transpose_cvt_kernel<<<g, 256, 0, stream>>>(wqkv, wqkvT, DIM, N3); }
  { dim3 g(DIM / 32, DIM / 32);
    transpose_cvt_kernel<<<g, 256, 0, stream>>>(wproj, wprojT, DIM, DIM); }

  dim3 g1(MROWS / 128, N3 / 128);     // 192 x 18
  qkv_gemm_kernel<<<g1, 256, 0, stream>>>(xh, wqkvT, qh, kh, vT);

  attention_kernel<<<BATCH * NHEADS * 6, 128, 0, stream>>>(qh, kh, vT, ao);

  dim3 g3(MROWS / 128, DIM / 128);    // 192 x 6
  proj_gemm_kernel<<<g3, 256, 0, stream>>>(ao, wprojT, bproj, out);
}
